// DynamicPerceiverAttention_20426864460544
// MI455X (gfx1250) — matrix operand order repaired and checked by reading, not compiled, not run on hardware
//
#include <hip/hip_runtime.h>
#include <hip/hip_bf16.h>

// ---------------------------------------------------------------------------
// DynamicPerceiverAttention for gfx1250 (CDNA5, wave32, WMMA f16 16x16x32)
// B=8 T=4 N1=1024 N2=64 D=1024 H=16 DH=64 INNER=1024 NK=1025
// ---------------------------------------------------------------------------

typedef _Float16 f16;
typedef __attribute__((ext_vector_type(16))) _Float16 v16h;
typedef __attribute__((ext_vector_type(8)))  _Float16 v8h;
typedef __attribute__((ext_vector_type(8)))  float    v8f;

#define BT      32
#define DMODEL  1024
#define INNER   1024
#define NHEAD   16
#define DHEAD   64
#define N2TOK   64
#define NK      1025     // image tokens + 1 latent token
#define NKPAD   1056     // 33 * 32
#define NTILES  33
#define LN_EPS  1e-5f

#ifndef __has_builtin
#define __has_builtin(x) 0
#endif
#if __has_builtin(__builtin_amdgcn_tensor_load_to_lds) && \
    __has_builtin(__builtin_amdgcn_s_wait_tensorcnt)
#define HAVE_TDM 1
#else
#define HAVE_TDM 0
#endif

__device__ __forceinline__ v16h cat8h(v8h lo, v8h hi) {
    return __builtin_shufflevector(lo, hi, 0,1,2,3,4,5,6,7,8,9,10,11,12,13,14,15);
}

__device__ __forceinline__ v8f wmma_f16f32(v16h a, v16h b, v8f c) {
    // D = A(16x32 f16) * B(32x16 f16) + C(16x16 f32)
    return __builtin_amdgcn_wmma_f32_16x16x32_f16(false, a, false, b, (short)0, c, false, false);
}

#if HAVE_TDM
typedef unsigned int u32;
typedef __attribute__((ext_vector_type(4))) unsigned int u32x4;
typedef __attribute__((ext_vector_type(8))) int i32x8;
typedef __attribute__((ext_vector_type(4))) int i32x4;

// 2D tensor tile -> LDS via Tensor Data Mover (6-arg clang-23 form: groups
// 0..3 + extra 8-dword group + cpol).  tensor dims == tile dims (tile known
// in-bounds), data_size = 2 bytes, LDS row padding via D# pad fields
// (pad_interval code: rows of 2<<code dwords, pad_amount code: (code+1)
// dwords appended).  All descriptor words forced to SGPRs.
__device__ __forceinline__ void tdm_load_tile_f16(const void* gsrc, u32 lds_off,
                                                  u32 tdim0, u32 tdim1,
                                                  unsigned long long stride_elems,
                                                  u32 pad_int_code, u32 pad_amt_code) {
    unsigned long long ga = (unsigned long long)gsrc;
    u32x4 g0;
    g0[0] = (u32)__builtin_amdgcn_readfirstlane((int)1u);                 // count=1
    g0[1] = (u32)__builtin_amdgcn_readfirstlane((int)lds_off);            // lds_addr
    g0[2] = (u32)__builtin_amdgcn_readfirstlane((int)(u32)ga);            // global_addr lo
    g0[3] = (u32)__builtin_amdgcn_readfirstlane(
        (int)(((u32)(ga >> 32) & 0x01ffffffu) | (2u << 30)));             // addr hi | type=2
    u32 w0 = (1u << 16) | (1u << 20) | (pad_int_code << 22) | (pad_amt_code << 25);
    i32x8 g1;
    g1[0] = __builtin_amdgcn_readfirstlane((int)w0);                      // data_size=2B, pad_en
    g1[1] = __builtin_amdgcn_readfirstlane((int)((tdim0 & 0xffffu) << 16));          // tensor_dim0 lo
    g1[2] = __builtin_amdgcn_readfirstlane(
        (int)(((tdim0 >> 16) & 0xffffu) | ((tdim1 & 0xffffu) << 16)));    // dim0 hi | dim1 lo
    g1[3] = __builtin_amdgcn_readfirstlane(
        (int)(((tdim1 >> 16) & 0xffffu) | (tdim0 << 16)));                // dim1 hi | tile_dim0
    g1[4] = __builtin_amdgcn_readfirstlane((int)(tdim1 & 0xffffu));       // tile_dim1, tile_dim2=0
    g1[5] = __builtin_amdgcn_readfirstlane((int)(u32)(stride_elems & 0xffffffffull));
    g1[6] = __builtin_amdgcn_readfirstlane((int)(u32)((stride_elems >> 32) & 0xffffull));
    g1[7] = 0;
    i32x4 z4 = {0, 0, 0, 0};
    i32x8 z8 = {0, 0, 0, 0, 0, 0, 0, 0};
    __builtin_amdgcn_tensor_load_to_lds(g0, g1, z4, z4, z8, 0);
}
#endif

// ---------------------------------------------------------------------------
// Weight transpose + fp32 -> f16 : Wt[n*K + k] = W[k*N + n]
// ---------------------------------------------------------------------------
__global__ __launch_bounds__(256)
void wtrans_f16_kernel(const float* __restrict__ W, f16* __restrict__ Wt,
                       int N, int K) {
    long idx = (long)blockIdx.x * blockDim.x + threadIdx.x;
    if (idx >= (long)N * K) return;
    int n = (int)(idx / K);
    int k = (int)(idx % K);
    Wt[idx] = (f16)W[(long)k * N + n];
}

// ---------------------------------------------------------------------------
// Row LayerNorm (D=1024) fp32 -> f16.  One block (256 thr) per row.
// ---------------------------------------------------------------------------
__global__ __launch_bounds__(256)
void ln_f16_kernel(const float* __restrict__ src,
                   const float* __restrict__ g,
                   const float* __restrict__ bvec,
                   f16* __restrict__ dst, int tokPerBT, long dstBTRows,
                   f16* __restrict__ dst2, long dst2BTRows) {
    long row = blockIdx.x;
    int bt  = (int)(row / tokPerBT);
    int tok = (int)(row % tokPerBT);
    const float* xr = src + row * DMODEL;
    int tid = threadIdx.x;

    float vals[4];
    float s = 0.f, ss = 0.f;
    #pragma unroll
    for (int i = 0; i < 4; i++) {
        float v = xr[tid + i * 256];
        vals[i] = v; s += v; ss += v * v;
    }
    __shared__ float r1[256], r2[256];
    r1[tid] = s; r2[tid] = ss;
    __syncthreads();
    for (int off = 128; off > 0; off >>= 1) {
        if (tid < off) { r1[tid] += r1[tid + off]; r2[tid] += r2[tid + off]; }
        __syncthreads();
    }
    float mu  = r1[0] * (1.0f / DMODEL);
    float var = r2[0] * (1.0f / DMODEL) - mu * mu;
    float rs  = rsqrtf(var + LN_EPS);

    f16* dr  = dst + ((long)bt * dstBTRows + tok) * DMODEL;
    f16* dr2 = (dst2 != nullptr && tok == 0)
                 ? (dst2 + (long)bt * dst2BTRows * DMODEL) : nullptr;
    #pragma unroll
    for (int i = 0; i < 4; i++) {
        int c = tid + i * 256;
        float y = (vals[i] - mu) * rs * g[c] + bvec[c];
        dr[c] = (f16)y;
        if (dr2) dr2[c] = (f16)y;
    }
}

// ---------------------------------------------------------------------------
// Dynamic per-head weights: pooled(ln) -> relu(Wd1) -> softmax(Wd2).
// ---------------------------------------------------------------------------
__global__ __launch_bounds__(256)
void dyn_weights_kernel(const f16* __restrict__ ln16,
                        const float* __restrict__ Wd1, const float* __restrict__ bd1,
                        const float* __restrict__ Wd2, const float* __restrict__ bd2,
                        float* __restrict__ dwout) {
    int bt = blockIdx.x;
    int tid = threadIdx.x;
    __shared__ float pooled[DMODEL];
    __shared__ float hid[256];
    __shared__ float logits[NHEAD];

    const f16* L = ln16 + (long)bt * N2TOK * DMODEL;
    for (int c = tid; c < DMODEL; c += 256) {
        float s = 0.f;
        for (int r = 0; r < N2TOK; r++) s += (float)L[r * DMODEL + c];
        pooled[c] = s * (1.0f / N2TOK);
    }
    __syncthreads();
    {
        float s = bd1[tid];
        for (int k = 0; k < DMODEL; k++) s += pooled[k] * Wd1[(long)k * 256 + tid];
        hid[tid] = fmaxf(s, 0.0f);
    }
    __syncthreads();
    if (tid < NHEAD) {
        float s = bd2[tid];
        for (int k = 0; k < 256; k++) s += hid[k] * Wd2[k * NHEAD + tid];
        logits[tid] = s;
    }
    __syncthreads();
    if (tid == 0) {
        float mx = logits[0];
        for (int i = 1; i < NHEAD; i++) mx = fmaxf(mx, logits[i]);
        float sum = 0.f, e[NHEAD];
        for (int i = 0; i < NHEAD; i++) { e[i] = __expf(logits[i] - mx); sum += e[i]; }
        float inv = 1.0f / sum;
        for (int i = 0; i < NHEAD; i++) dwout[bt * NHEAD + i] = e[i] * inv;
    }
}

// ---------------------------------------------------------------------------
// Tiled WMMA GEMM:  C[M,N] = alpha * A[M,K] * Bt[N,K]^T    (A,Bt f16)
// 256 thr = 8 waves -> 128x128 tile, wave = 64x32 (4x2 wmma), K-step 32,
// double-buffered LDS (1 barrier/step) + L2 prefetch 2 steps ahead.
// MODE 0: C0 f16 row-major | MODE 1: C0 f32 row-major
// MODE 2: kv split: col<INNER -> C0 f16 [row][col] ; else C1 f16 [col-INNER][row]
// ---------------------------------------------------------------------------
template<int MODE>
__global__ __launch_bounds__(256)
void gemm_wmma_kernel(const f16* __restrict__ Abase, const f16* __restrict__ Bt,
                      void* __restrict__ C0, void* __restrict__ C1,
                      int M, int N, int K,
                      long aBT, long c0BT, long c1BT,
                      int ldc0, int ldc1, float alpha) {
    const int bt = blockIdx.z;
    const f16* A = Abase + (long)bt * aBT;
    const int n0 = blockIdx.x * 128;
    const int m0 = blockIdx.y * 128;

    __shared__ f16 Als[2][128][40];   // ping-pong 128x32 (+pad)
    __shared__ f16 Bls[2][128][40];

    const int tid  = threadIdx.x;
    const int lane = tid & 31;
    const int wid  = tid >> 5;
    const int wm   = (wid & 1) * 64;
    const int wn   = (wid >> 1) * 32;
    const int half = lane >> 4;
    const int lm   = lane & 15;

    v8f acc[4][2] = {};

    auto load_tile = [&](int k0, int buf) {
        for (int c = tid; c < 1024; c += 256) {
            int cc  = c & 511;
            int row = cc >> 2;
            int co  = (cc & 3) * 8;
            uint4 v = {0u, 0u, 0u, 0u};
            if (c < 512) {
                int gr = m0 + row;
                if (gr < M) v = *(const uint4*)(A + (long)gr * K + k0 + co);
                *(uint4*)&Als[buf][row][co] = v;
            } else {
                int gn = n0 + row;
                if (gn < N) v = *(const uint4*)(Bt + (long)gn * K + k0 + co);
                *(uint4*)&Bls[buf][row][co] = v;
            }
        }
    };

    load_tile(0, 0);
    __syncthreads();

    const int nk = K >> 5;
    for (int it = 0; it < nk; ++it) {
        const int cur = it & 1;
        if (it + 1 < nk) {
            load_tile((it + 1) << 5, cur ^ 1);
            if (it + 2 < nk) {     // prefetch 2 steps ahead into L2/WGP$
                int kpf = (it + 2) << 5;
                if (tid < 128) {
                    int gr = m0 + tid;
                    if (gr < M) __builtin_prefetch(A + (long)gr * K + kpf, 0, 1);
                } else {
                    int gn = n0 + (tid - 128);
                    if (gn < N) __builtin_prefetch(Bt + (long)gn * K + kpf, 0, 1);
                }
            }
        }

        v16h afrag[4], bfrag[2];
        #pragma unroll
        for (int mi = 0; mi < 4; mi++) {
            const f16* ar = &Als[cur][wm + mi * 16 + lm][0];
            v8h lo = *(const v8h*)(ar + half * 8);
            v8h hi = *(const v8h*)(ar + 16 + half * 8);
            afrag[mi] = cat8h(lo, hi);
        }
        #pragma unroll
        for (int ni = 0; ni < 2; ni++) {
            const f16* br = &Bls[cur][wn + ni * 16 + lm][half * 8];
            v8h lo = *(const v8h*)br;
            v8h hi = *(const v8h*)(br + 16);
            bfrag[ni] = cat8h(lo, hi);
        }
        #pragma unroll
        for (int mi = 0; mi < 4; mi++)
            #pragma unroll
            for (int ni = 0; ni < 2; ni++)
                acc[mi][ni] = wmma_f16f32(afrag[mi], bfrag[ni], acc[mi][ni]);
        __syncthreads();
    }

    #pragma unroll
    for (int mi = 0; mi < 4; mi++) {
        #pragma unroll
        for (int ni = 0; ni < 2; ni++) {
            #pragma unroll
            for (int j = 0; j < 8; j++) {
                int r = m0 + wm + mi * 16 + half * 8 + j;
                int c = n0 + wn + ni * 16 + lm;
                if (r >= M) continue;
                float v = acc[mi][ni][j] * alpha;
                if (MODE == 0) {
                    ((f16*)C0)[(long)bt * c0BT + (long)r * ldc0 + c] = (f16)v;
                } else if (MODE == 1) {
                    ((float*)C0)[(long)bt * c0BT + (long)r * ldc0 + c] = v;
                } else {
                    if (c < INNER)
                        ((f16*)C0)[(long)bt * c0BT + (long)r * ldc0 + c] = (f16)v;
                    else
                        ((f16*)C1)[(long)bt * c1BT + (long)(c - INNER) * ldc1 + r] = (f16)v;
                }
            }
        }
    }
}

// ---------------------------------------------------------------------------
// Flash attention, one wave per (bt, head).  K/V tiles staged in LDS by the
// Tensor Data Mover (double-buffered, TENSORcnt-synchronized, LDS row pad via
// D# pad fields) when available; falls back to direct global fragment loads.
// ---------------------------------------------------------------------------
#define KROWH 72   // 64 halves + 8 pad (TDM pad: 32-dword rows + 4 dwords)
#define VROWH 40   // 32 halves + 8 pad (TDM pad: 16-dword rows + 4 dwords)

__global__ __launch_bounds__(32)
void attn_flash_kernel(const f16* __restrict__ qbuf, const f16* __restrict__ kbuf,
                       const f16* __restrict__ vT, const float* __restrict__ dw,
                       f16* __restrict__ obuf) {
    const int bt = blockIdx.x;
    const int h  = blockIdx.y;
    const int lane = threadIdx.x & 31;
    const int half = lane >> 4;
    const int lm   = lane & 15;

    const f16* q  = qbuf + (long)bt * 64 * INNER + h * DHEAD;
    const f16* kp = kbuf + (long)bt * NKPAD * INNER + h * DHEAD;
    const f16* vp = vT   + ((long)bt * INNER + h * DHEAD) * NKPAD;

    v16h qf[4][2];
    #pragma unroll
    for (int mi = 0; mi < 4; mi++)
        #pragma unroll
        for (int ks = 0; ks < 2; ks++) {
            const f16* qr = q + (long)(mi * 16 + lm) * INNER + ks * 32;
            v8h lo = *(const v8h*)(qr + half * 8);
            v8h hi = *(const v8h*)(qr + 16 + half * 8);
            qf[mi][ks] = cat8h(lo, hi);
        }

    v16h ones;
    #pragma unroll
    for (int i = 0; i < 16; i++) ones[i] = (f16)1.0f;

    v8f O[4][4] = {};
    float mrun[4][8], lrun[4][8];
    #pragma unroll
    for (int mi = 0; mi < 4; mi++)
        #pragma unroll
        for (int j = 0; j < 8; j++) { mrun[mi][j] = -1e30f; lrun[mi][j] = 0.0f; }

    __shared__ f16 Pls[64][40];
#if HAVE_TDM
    __shared__ __align__(16) f16 Kls[2][32 * KROWH];
    __shared__ __align__(16) f16 Vls[2][64 * VROWH];
    // prime tile 0: K tile 32 tok x 64 dh (row stride INNER elems),
    //               V tile 64 dcol x 32 tok (row stride NKPAD elems)
    tdm_load_tile_f16(kp, (u32)(unsigned long long)(const void*)&Kls[0][0],
                      64, 32, (unsigned long long)INNER, 4, 3);
    tdm_load_tile_f16(vp, (u32)(unsigned long long)(const void*)&Vls[0][0],
                      32, 64, (unsigned long long)NKPAD, 3, 3);
#endif

    for (int it = 0; it < NTILES; ++it) {
        const int t0 = it * 32;
#if HAVE_TDM
        const int cur = it & 1;
        if (it + 1 < NTILES) {
            // previous reads of the other buffer must have landed before the
            // TDM overwrites it
            asm volatile("s_wait_dscnt 0" ::: "memory");
            tdm_load_tile_f16(kp + (long)(t0 + 32) * INNER,
                              (u32)(unsigned long long)(const void*)&Kls[cur ^ 1][0],
                              64, 32, (unsigned long long)INNER, 4, 3);
            tdm_load_tile_f16(vp + (t0 + 32),
                              (u32)(unsigned long long)(const void*)&Vls[cur ^ 1][0],
                              32, 64, (unsigned long long)NKPAD, 3, 3);
            __builtin_amdgcn_s_wait_tensorcnt(2);   // current pair complete
        } else {
            __builtin_amdgcn_s_wait_tensorcnt(0);
        }
        const f16* Kb = &Kls[cur][0];
        const f16* Vb = &Vls[cur][0];
#endif
        // K^T fragments: [ks][ni], lane lm = token lane, K = 32*ks+8*half+e (e<8), 32*ks+16+8*half+(e-8)
        v16h kf[2][2];
        #pragma unroll
        for (int ni = 0; ni < 2; ni++)
            #pragma unroll
            for (int ks = 0; ks < 2; ks++) {
#if HAVE_TDM
                const f16* kr = Kb + (ni * 16 + lm) * KROWH + ks * 32 + half * 8;
#else
                const f16* kr = kp + (long)(t0 + ni * 16 + lm) * INNER + ks * 32 + half * 8;
#endif
                v8h lo = *(const v8h*)kr;
                v8h hi = *(const v8h*)(kr + 16);
                kf[ks][ni] = cat8h(lo, hi);
            }

        v8f S[4][2];
        #pragma unroll
        for (int mi = 0; mi < 4; mi++)
            #pragma unroll
            for (int ni = 0; ni < 2; ni++) {
                v8f s = {};
                s = wmma_f16f32(qf[mi][0], kf[0][ni], s);
                s = wmma_f16f32(qf[mi][1], kf[1][ni], s);
                S[mi][ni] = s;
            }

        // mask padded tokens (valid: 0..NK-1)
        #pragma unroll
        for (int ni = 0; ni < 2; ni++) {
            int token = t0 + ni * 16 + lm;
            if (token >= NK) {
                #pragma unroll
                for (int mi = 0; mi < 4; mi++)
                    #pragma unroll
                    for (int j = 0; j < 8; j++) S[mi][ni][j] = -1e30f;
            }
        }

        // online softmax update + stage P into LDS (row = 16mi+8half+j)
        #pragma unroll
        for (int mi = 0; mi < 4; mi++) {
            #pragma unroll
            for (int j = 0; j < 8; j++) {
                float sm = fmaxf(S[mi][0][j], S[mi][1][j]);
                sm = fmaxf(sm, __shfl_xor(sm, 8, 32));
                sm = fmaxf(sm, __shfl_xor(sm, 4, 32));
                sm = fmaxf(sm, __shfl_xor(sm, 2, 32));
                sm = fmaxf(sm, __shfl_xor(sm, 1, 32));
                float mnew = fmaxf(mrun[mi][j], sm);
                float corr = __expf(mrun[mi][j] - mnew);
                mrun[mi][j] = mnew;
                lrun[mi][j] *= corr;
                #pragma unroll
                for (int di = 0; di < 4; di++) O[mi][di][j] *= corr;
                S[mi][0][j] = __expf(S[mi][0][j] - mnew);
                S[mi][1][j] = __expf(S[mi][1][j] - mnew);
            }
            #pragma unroll
            for (int ni = 0; ni < 2; ni++)
                #pragma unroll
                for (int j = 0; j < 8; j++)
                    Pls[mi * 16 + half * 8 + j][ni * 16 + lm] = (f16)S[mi][ni][j];
        }
        __syncthreads();

        // P fragments (A, 16x32 over 32 tokens) and V fragments (B)
        v16h pf[4], vf[4];
        #pragma unroll
        for (int mi = 0; mi < 4; mi++) {
            const f16* pr = &Pls[mi * 16 + lm][0];
            v8h lo = *(const v8h*)(pr + half * 8);
            v8h hi = *(const v8h*)(pr + 16 + half * 8);
            pf[mi] = cat8h(lo, hi);
        }
        #pragma unroll
        for (int di = 0; di < 4; di++) {
#if HAVE_TDM
            const f16* vr = Vb + (di * 16 + lm) * VROWH + half * 8;
#else
            const f16* vr = vp + (long)(di * 16 + lm) * NKPAD + t0 + half * 8;
#endif
            v8h lo = *(const v8h*)vr;
            v8h hi = *(const v8h*)(vr + 16);
            vf[di] = cat8h(lo, hi);
        }

        #pragma unroll
        for (int mi = 0; mi < 4; mi++) {
            v8f lacc = {};
            lacc = wmma_f16f32(pf[mi], ones, lacc);   // row-sums via WMMA
            #pragma unroll
            for (int j = 0; j < 8; j++) lrun[mi][j] += lacc[j];
            #pragma unroll
            for (int di = 0; di < 4; di++)
                O[mi][di] = wmma_f16f32(pf[mi], vf[di], O[mi][di]);
        }
        __syncthreads();
    }

    const float w = dw[bt * NHEAD + h];
    f16* op = obuf + (long)bt * 64 * INNER + h * DHEAD;
    #pragma unroll
    for (int mi = 0; mi < 4; mi++)
        #pragma unroll
        for (int di = 0; di < 4; di++)
            #pragma unroll
            for (int j = 0; j < 8; j++) {
                int r = mi * 16 + half * 8 + j;
                int c = di * 16 + lm;
                op[(long)r * INNER + c] = (f16)(O[mi][di][j] * w / lrun[mi][j]);
            }
}

// ---------------------------------------------------------------------------
// Host launcher
// ---------------------------------------------------------------------------
extern "C" void kernel_launch(void* const* d_in, const int* in_sizes, int n_in,
                              void* d_out, int out_size, void* d_ws, size_t ws_size,
                              hipStream_t stream) {
    const float* x       = (const float*)d_in[0];
    const float* latents = (const float*)d_in[1];
    const float* gm      = (const float*)d_in[2];
    const float* bm      = (const float*)d_in[3];
    const float* gl      = (const float*)d_in[4];
    const float* bl      = (const float*)d_in[5];
    const float* Wq      = (const float*)d_in[6];
    const float* Wkv     = (const float*)d_in[7];
    const float* Wout    = (const float*)d_in[8];
    const float* Wd1     = (const float*)d_in[9];
    const float* bd1     = (const float*)d_in[10];
    const float* Wd2     = (const float*)d_in[11];
    const float* bd2     = (const float*)d_in[12];
    float* out = (float*)d_out;

    char* p = (char*)d_ws;
    auto carve = [&](size_t bytes) -> void* {
        void* r = (void*)p;
        p += (bytes + 255) & ~(size_t)255;
        return r;
    };
    f16* kvin   = (f16*)carve((size_t)BT * NK * DMODEL * sizeof(f16));
    f16* ln16   = (f16*)carve((size_t)BT * N2TOK * DMODEL * sizeof(f16));
    f16* Wq_t   = (f16*)carve((size_t)INNER * DMODEL * sizeof(f16));
    f16* Wkv_t  = (f16*)carve((size_t)2 * INNER * DMODEL * sizeof(f16));
    f16* Wout_t = (f16*)carve((size_t)DMODEL * INNER * sizeof(f16));
    f16* qbuf   = (f16*)carve((size_t)BT * N2TOK * INNER * sizeof(f16));
    f16* kbuf   = (f16*)carve((size_t)BT * NKPAD * INNER * sizeof(f16));
    f16* vTbuf  = (f16*)carve((size_t)BT * INNER * NKPAD * sizeof(f16));
    f16* aobuf  = (f16*)carve((size_t)BT * N2TOK * INNER * sizeof(f16));
    float* dwbuf= (float*)carve((size_t)BT * NHEAD * sizeof(float));

    wtrans_f16_kernel<<<(INNER * DMODEL) / 256, 256, 0, stream>>>(Wq,   Wq_t,   INNER,     DMODEL);
    wtrans_f16_kernel<<<(2 * INNER * DMODEL) / 256, 256, 0, stream>>>(Wkv, Wkv_t, 2 * INNER, DMODEL);
    wtrans_f16_kernel<<<(DMODEL * INNER) / 256, 256, 0, stream>>>(Wout, Wout_t, DMODEL,    INNER);

    ln_f16_kernel<<<BT * 1024, 256, 0, stream>>>(x, gm, bm, kvin, 1024, NK, nullptr, 0);
    ln_f16_kernel<<<BT * N2TOK, 256, 0, stream>>>(latents, gl, bl, ln16, N2TOK, N2TOK,
                                                  kvin + (size_t)(NK - 1) * DMODEL, NK);

    dyn_weights_kernel<<<BT, 256, 0, stream>>>(ln16, Wd1, bd1, Wd2, bd2, dwbuf);

    {   // kv = kvin @ Wkv  (M=1025, N=2048, K=1024) -> k row-major + vT
        dim3 grid(2 * INNER / 128, (NK + 127) / 128, BT);
        gemm_wmma_kernel<2><<<grid, 256, 0, stream>>>(
            kvin, Wkv_t, (void*)kbuf, (void*)vTbuf,
            NK, 2 * INNER, DMODEL,
            (long)NK * DMODEL, (long)NKPAD * INNER, (long)INNER * NKPAD,
            INNER, NKPAD, 1.0f);
    }
    {   // q = ln @ Wq (scale dh^-0.5 folded)
        dim3 grid(INNER / 128, 1, BT);
        gemm_wmma_kernel<0><<<grid, 256, 0, stream>>>(
            ln16, Wq_t, (void*)qbuf, nullptr,
            N2TOK, INNER, DMODEL,
            (long)N2TOK * DMODEL, (long)N2TOK * INNER, 0,
            INNER, 0, 0.125f);
    }
    {   // flash attention
        dim3 grid(BT, NHEAD);
        attn_flash_kernel<<<grid, 32, 0, stream>>>(qbuf, kbuf, vTbuf, dwbuf, aobuf);
    }
    {   // out = attn_out @ Wout -> f32
        dim3 grid(DMODEL / 128, 1, BT);
        gemm_wmma_kernel<1><<<grid, 256, 0, stream>>>(
            aobuf, Wout_t, (void*)out, nullptr,
            N2TOK, DMODEL, INNER,
            (long)N2TOK * INNER, (long)N2TOK * DMODEL, 0,
            DMODEL, 0, 1.0f);
    }
}
